// AsymmeUpBlock_3152505995484
// MI455X (gfx1250) — matrix operand order repaired and checked by reading, not compiled, not run on hardware
//
#include <hip/hip_runtime.h>

typedef __attribute__((ext_vector_type(16))) _Float16 v16h;
typedef __attribute__((ext_vector_type(8)))  _Float16 v8h;
typedef __attribute__((ext_vector_type(8)))  float    v8f;

// ---------------------------------------------------------------------------
// Weight pack: src (Cout, Cin, KD,KH,KW) f32  ->  dst[tap][cout][cin] f16
// ---------------------------------------------------------------------------
__global__ void pack_weights_kernel(const float* __restrict__ src,
                                    _Float16* __restrict__ dst,
                                    int cout, int cin, int taps) {
  int i = blockIdx.x * blockDim.x + threadIdx.x;
  int total = cout * cin * taps;
  if (i >= total) return;
  int tap = i % taps;
  int t   = i / taps;
  int ci  = t % cin;
  int co  = t / cin;
  dst[((size_t)tap * cout + co) * cin + ci] = (_Float16)src[i];
}

// f32 -> f16 elementwise (used once for the network input)
__global__ void f32_to_f16_kernel(const float* __restrict__ x,
                                  _Float16* __restrict__ y, int n) {
  int i = blockIdx.x * blockDim.x + threadIdx.x;
  if (i < n) y[i] = (_Float16)x[i];
}

// ---------------------------------------------------------------------------
// Implicit-GEMM conv via WMMA f16 -> f32.
//   M = 64 output channels (4 waves x 16), N = 16 spatial W positions / block,
//   K = Cin * taps, marched 32 at a time (one tap x 32 input channels).
// A (weights) staged to LDS with GLOBAL_LOAD_ASYNC_TO_LDS_B128 (ASYNCcnt).
// TRANS=true implements the stride-2 transposed conv (gather form, pad_lo=1).
// ---------------------------------------------------------------------------
template <int KD, int KH, int KW, int PD, int PH, int PW, int CIN,
          bool TRANS, bool ADD_SKIP, bool LRELU, typename OutT>
__global__ __launch_bounds__(128)
void conv_wmma_kernel(const _Float16* __restrict__ in,
                      const _Float16* __restrict__ wp,
                      const float* __restrict__ skip,
                      OutT* __restrict__ out,
                      int ID, int IH, int IW,
                      int OD, int OH, int OW) {
  constexpr int TAPS = KD * KH * KW;
  __shared__ __align__(16) _Float16 As[64 * 32];   // [cout][k]  4 KB
  __shared__ __align__(16) _Float16 Bs[16 * 32];   // [x][k]     1 KB

  const int ox0  = blockIdx.x * 16;
  const int oy   = blockIdx.y;
  const int oz   = blockIdx.z;
  const int tid  = threadIdx.x;
  const int lane = tid & 31;
  const int wv   = tid >> 5;     // wave 0..3
  const int m0   = wv * 16;      // cout tile base

  v8f acc = {0.f, 0.f, 0.f, 0.f, 0.f, 0.f, 0.f, 0.f};

  for (int tap = 0; tap < TAPS; ++tap) {
    const int kw = tap % KW;
    const int kh = (tap / KW) % KH;
    const int kd = tap / (KW * KH);

    int iz, iy;
    bool zyok;
    if (TRANS) {
      int tz = oz + kd - 1, ty = oy + kh - 1;
      zyok = ((tz & 1) == 0) && ((ty & 1) == 0);
      iz = tz >> 1;
      iy = ty >> 1;
      zyok = zyok && (unsigned)iz < (unsigned)ID && (unsigned)iy < (unsigned)IH;
    } else {
      iz = oz + kd - PD;
      iy = oy + kh - PH;
      zyok = (unsigned)iz < (unsigned)ID && (unsigned)iy < (unsigned)IH;
    }

    for (int c0 = 0; c0 < CIN; c0 += 32) {
      // ---- stage A: 64 couts x 32 k halves, async copy (32 B / thread).
      //      Instruction's immediate offset applies to BOTH lds & global addr.
      {
        const int r = tid >> 1;            // cout row
        const int h = (tid & 1) * 16;      // 16-half chunk
        const _Float16* g = wp + ((size_t)(tap * 64 + r)) * CIN + c0 + h;
        unsigned lds_a = (unsigned)(uintptr_t)(&As[r * 32 + h]);
        asm volatile(
            "global_load_async_to_lds_b128 %0, %1, off\n\t"
            "global_load_async_to_lds_b128 %0, %1, off offset:16"
            :: "v"(lds_a), "v"(g) : "memory");
      }
      // ---- stage B: 16 x positions x 32 cin halves, zero-filled OOB
      {
        const int cl = tid >> 2;           // cin local 0..31
        const int xg = tid & 3;            // group of 4 x positions
        const _Float16* cp =
            in + (((size_t)(c0 + cl) * ID + iz) * IH + iy) * IW;
        _Float16 v[4];
#pragma unroll
        for (int i = 0; i < 4; ++i) {
          int n = xg * 4 + i;
          _Float16 val = (_Float16)0.f;
          if (zyok) {
            if (TRANS) {
              int tx = ox0 + n + kw - 1;
              if ((tx & 1) == 0) {
                int ix = tx >> 1;
                if ((unsigned)ix < (unsigned)IW) val = cp[ix];
              }
            } else {
              int ix = ox0 + n + kw - PW;
              if ((unsigned)ix < (unsigned)IW) val = cp[ix];
            }
          }
          v[i] = val;
        }
#pragma unroll
        for (int i = 0; i < 4; ++i)
          Bs[(xg * 4 + i) * 32 + cl] = v[i];
      }
      // Wait for this wave's async LDS writes, then sync the block.
      asm volatile("s_wait_asynccnt 0x0" ::: "memory");
      __syncthreads();

      // ---- A fragment (ISA 7.12.2 16-bit A 16x32):
      //   M = lane&15; lanes 0-15: K in {0..7,16..23}; lanes 16-31: +8
      const int mr    = m0 + (lane & 15);
      const int ahalf = (lane >> 4) * 8;
      v8h alo = *(const v8h*)(&As[mr * 32 + ahalf]);
      v8h ahi = *(const v8h*)(&As[mr * 32 + ahalf + 16]);
      v16h afrag = __builtin_shufflevector(alo, ahi, 0, 1, 2, 3, 4, 5, 6, 7,
                                           8, 9, 10, 11, 12, 13, 14, 15);
      // ---- B fragment: N = lane&15; lanes 0-15 K in {0..7,16..23}; lanes 16-31: +8
      const int nx = lane & 15;
      const int bk = (lane >> 4) * 8;
      v8h blo = *(const v8h*)(&Bs[nx * 32 + bk]);
      v8h bhi = *(const v8h*)(&Bs[nx * 32 + bk + 16]);
      v16h bfrag = __builtin_shufflevector(blo, bhi, 0, 1, 2, 3, 4, 5, 6, 7,
                                           8, 9, 10, 11, 12, 13, 14, 15);

      acc = __builtin_amdgcn_wmma_f32_16x16x32_f16(
          /*neg_a=*/false, afrag, /*neg_b=*/false, bfrag,
          /*c_mod=*/(short)0, acc, /*reuse_a=*/false, /*reuse_b=*/false);
      __syncthreads();
    }
  }

  // ---- epilogue (C layout: N=lane&15; VGPR r -> M = r + 8*(lane>=16))
  const int n  = lane & 15;
  const int mh = (lane >> 4) * 8;
#pragma unroll
  for (int r = 0; r < 8; ++r) {
    int m = m0 + mh + r;
    float vo = acc[r];
    if (LRELU) vo = vo >= 0.f ? vo : 0.01f * vo;
    size_t idx = (((size_t)m * OD + oz) * OH + oy) * OW + ox0 + n;
    if (ADD_SKIP) vo += skip[idx];
    out[idx] = (OutT)vo;
  }
}

// ---------------------------------------------------------------------------
// BatchNorm helpers (per-channel stats over D*H*W, two-pass)
// ---------------------------------------------------------------------------
__global__ void zero_stats_kernel(float* s) {
  if (threadIdx.x < 128) s[threadIdx.x] = 0.f;
}

__global__ __launch_bounds__(256)
void bn_stats_kernel(const float* __restrict__ x, float* __restrict__ stats,
                     int nper) {
  int c = blockIdx.y;
  const float* p = x + (size_t)c * nper;
  float s = 0.f, ss = 0.f;
  for (int i = blockIdx.x * blockDim.x + threadIdx.x; i < nper;
       i += gridDim.x * blockDim.x) {
    float v = p[i];
    s += v;
    ss += v * v;
  }
  __shared__ float sh0[256];
  __shared__ float sh1[256];
  sh0[threadIdx.x] = s;
  sh1[threadIdx.x] = ss;
  __syncthreads();
  for (int o = 128; o > 0; o >>= 1) {
    if ((int)threadIdx.x < o) {
      sh0[threadIdx.x] += sh0[threadIdx.x + o];
      sh1[threadIdx.x] += sh1[threadIdx.x + o];
    }
    __syncthreads();
  }
  if (threadIdx.x == 0) {
    atomicAdd(&stats[2 * c], sh0[0]);
    atomicAdd(&stats[2 * c + 1], sh1[0]);
  }
}

// Normalize f32 pre-BN buffer, write f16 (feeds the next conv's B staging).
__global__ __launch_bounds__(256)
void bn_apply_f16_kernel(const float* __restrict__ x, _Float16* __restrict__ y,
                         const float* __restrict__ stats,
                         const float* __restrict__ g,
                         const float* __restrict__ b, int nper) {
  int c = blockIdx.y;
  float inv_n = 1.0f / (float)nper;
  float mean = stats[2 * c] * inv_n;
  float var  = stats[2 * c + 1] * inv_n - mean * mean;
  float sc = rsqrtf(var + 1e-5f) * g[c];
  float sh = b[c] - mean * sc;
  const float* p = x + (size_t)c * nper;
  _Float16* q = y + (size_t)c * nper;
  for (int i = blockIdx.x * blockDim.x + threadIdx.x; i < nper;
       i += gridDim.x * blockDim.x)
    q[i] = (_Float16)(p[i] * sc + sh);
}

// Normalize f32 in place (final output).
__global__ __launch_bounds__(256)
void bn_apply_kernel(float* __restrict__ x, const float* __restrict__ stats,
                     const float* __restrict__ g, const float* __restrict__ b,
                     int nper) {
  int c = blockIdx.y;
  float inv_n = 1.0f / (float)nper;
  float mean = stats[2 * c] * inv_n;
  float var  = stats[2 * c + 1] * inv_n - mean * mean;
  float sc = rsqrtf(var + 1e-5f) * g[c];
  float sh = b[c] - mean * sc;
  float* p = x + (size_t)c * nper;
  for (int i = blockIdx.x * blockDim.x + threadIdx.x; i < nper;
       i += gridDim.x * blockDim.x)
    p[i] = p[i] * sc + sh;
}

// ---------------------------------------------------------------------------
// Host-side orchestration
// ---------------------------------------------------------------------------
static inline size_t align_up(size_t v, size_t a) { return (v + a - 1) & ~(a - 1); }

extern "C" void kernel_launch(void* const* d_in, const int* in_sizes, int n_in,
                              void* d_out, int out_size, void* d_ws, size_t ws_size,
                              hipStream_t stream) {
  (void)in_sizes; (void)n_in; (void)out_size; (void)ws_size;
  const float* x       = (const float*)d_in[0];   // (1,128,16,48,48)
  const float* skip    = (const float*)d_in[1];   // (1,64,32,96,96)
  const float* w_trans = (const float*)d_in[2];   // (64,128,3,3,3)
  const float* g_t     = (const float*)d_in[3];
  const float* b_t     = (const float*)d_in[4];
  const float* w_up    = (const float*)d_in[5];   // (64,64,3,3,3)
  const float* w1      = (const float*)d_in[6];   // (64,64,1,3,3)
  const float* g1      = (const float*)d_in[7];
  const float* b1      = (const float*)d_in[8];
  const float* w2      = (const float*)d_in[9];   // (64,64,3,1,3)
  const float* g2      = (const float*)d_in[10];
  const float* b2      = (const float*)d_in[11];
  const float* w3      = (const float*)d_in[12];  // (64,64,3,3,3)
  const float* g3      = (const float*)d_in[13];
  const float* b3      = (const float*)d_in[14];
  float* outp = (float*)d_out;                    // (1,64,32,96,96)

  // ---- workspace carve-up
  char* ws = (char*)d_ws;
  size_t off = 0;
  auto carve = [&](size_t bytes) {
    size_t o = off;
    off = align_up(off + bytes, 256);
    return (void*)(ws + o);
  };
  const size_t NCOARSE = (size_t)64 * 16 * 48 * 48;   // 2.36M
  const size_t NFINE   = (size_t)64 * 32 * 96 * 96;   // 18.9M
  _Float16* wtr = (_Float16*)carve((size_t)64 * 128 * 27 * 2);
  _Float16* wup = (_Float16*)carve((size_t)64 * 64 * 27 * 2);
  _Float16* w1p = (_Float16*)carve((size_t)64 * 64 * 9 * 2);
  _Float16* w2p = (_Float16*)carve((size_t)64 * 64 * 9 * 2);
  _Float16* w3p = (_Float16*)carve((size_t)64 * 64 * 27 * 2);
  float*    stats = (float*)carve(128 * sizeof(float));
  _Float16* hX = (_Float16*)carve((size_t)128 * 16 * 48 * 48 * 2); // f16 input
  _Float16* hA = (_Float16*)carve(NCOARSE * 2);  // f16 coarse activ.
  _Float16* hB = (_Float16*)carve(NFINE * 2);    // f16 fine activ.
  _Float16* hC = (_Float16*)carve(NFINE * 2);    // f16 fine activ.
  float* t32c = (float*)carve(NCOARSE * 4);      // f32 pre-BN (coarse)
  float* t32f = (float*)carve(NFINE * 4);        // f32 pre-BN (fine)

  // ---- pack weights to f16, [tap][cout][cin]; convert input to f16
  auto pack = [&](const float* src, _Float16* dst, int cout, int cin, int taps) {
    int total = cout * cin * taps;
    pack_weights_kernel<<<(total + 255) / 256, 256, 0, stream>>>(src, dst, cout,
                                                                 cin, taps);
  };
  pack(w_trans, wtr, 64, 128, 27);
  pack(w_up,   wup, 64, 64, 27);
  pack(w1,     w1p, 64, 64, 9);
  pack(w2,     w2p, 64, 64, 9);
  pack(w3,     w3p, 64, 64, 27);
  {
    int n = 128 * 16 * 48 * 48;
    f32_to_f16_kernel<<<(n + 255) / 256, 256, 0, stream>>>(x, hX, n);
  }

  const int NC = 36864;    // 16*48*48
  const int NF = 294912;   // 32*96*96

  auto bn_to_f16 = [&](const float* src, _Float16* dst, const float* g,
                       const float* b, int nper) {
    zero_stats_kernel<<<1, 128, 0, stream>>>(stats);
    bn_stats_kernel<<<dim3(32, 64), 256, 0, stream>>>(src, stats, nper);
    bn_apply_f16_kernel<<<dim3(32, 64), 256, 0, stream>>>(src, dst, stats, g, b,
                                                          nper);
  };

  // 1) trans conv 3x3x3 (128->64) + lrelu -> f32, BN -> f16
  conv_wmma_kernel<3, 3, 3, 1, 1, 1, 128, false, false, true, float>
      <<<dim3(3, 48, 16), 128, 0, stream>>>(hX, wtr, nullptr, t32c,
                                            16, 48, 48, 16, 48, 48);
  bn_to_f16(t32c, hA, g_t, b_t, NC);

  // 2) transposed conv (stride-2 upsample) + skip -> f16 directly
  conv_wmma_kernel<3, 3, 3, 1, 1, 1, 64, true, true, false, _Float16>
      <<<dim3(6, 96, 32), 128, 0, stream>>>(hA, wup, skip, hB,
                                            16, 48, 48, 32, 96, 96);

  // 3) conv (1,3,3) + lrelu -> f32, BN -> f16
  conv_wmma_kernel<1, 3, 3, 0, 1, 1, 64, false, false, true, float>
      <<<dim3(6, 96, 32), 128, 0, stream>>>(hB, w1p, nullptr, t32f,
                                            32, 96, 96, 32, 96, 96);
  bn_to_f16(t32f, hC, g1, b1, NF);

  // 4) conv (3,1,3) + lrelu -> f32, BN -> f16 (reuse hB)
  conv_wmma_kernel<3, 1, 3, 1, 0, 1, 64, false, false, true, float>
      <<<dim3(6, 96, 32), 128, 0, stream>>>(hC, w2p, nullptr, t32f,
                                            32, 96, 96, 32, 96, 96);
  bn_to_f16(t32f, hB, g2, b2, NF);

  // 5) conv (3,3,3) + lrelu -> d_out (f32), BN in place
  conv_wmma_kernel<3, 3, 3, 1, 1, 1, 64, false, false, true, float>
      <<<dim3(6, 96, 32), 128, 0, stream>>>(hB, w3p, nullptr, outp,
                                            32, 96, 96, 32, 96, 96);
  zero_stats_kernel<<<1, 128, 0, stream>>>(stats);
  bn_stats_kernel<<<dim3(32, 64), 256, 0, stream>>>(outp, stats, NF);
  bn_apply_kernel<<<dim3(32, 64), 256, 0, stream>>>(outp, stats, g3, b3, NF);
}
